// CharacterDelimitedAttention_35218731827802
// MI455X (gfx1250) — matrix operand order repaired and checked by reading, not compiled, not run on hardware
//
#include <hip/hip_runtime.h>

typedef __bf16 bf16;
typedef __attribute__((ext_vector_type(16))) __bf16 v16bf;
typedef __attribute__((ext_vector_type(8)))  float  v8f;

#define B_  4
#define S_  2048
#define E_  1024
#define H_  16
#define D_  64
#define N3_ (3 * E_)
#define QT_ 128
#define BS_ (B_ * S_)

struct alignas(16) U4 { unsigned int x, y, z, w; };
struct alignas(16) F4 { float x, y, z, w; };
struct alignas(8)  B4 { bf16 h[4]; };

union FragU { v16bf v; U4 u[2]; };

// A-matrix fragment (16 rows in lanes): lane L holds row (L&15);
// elements e=0..7 -> K = 8*half + e, e=8..15 -> K = 16 + 8*half + (e-8)
__device__ __forceinline__ v16bf load_frag_a(const bf16* base, int stride, int lane) {
  const int r = lane & 15, hh = lane >> 4;
  const bf16* p = base + (size_t)r * stride + 8 * hh;
  FragU f;
  f.u[0] = *reinterpret_cast<const U4*>(p);
  f.u[1] = *reinterpret_cast<const U4*>(p + 16);
  return f.v;
}

// B-matrix fragment (16 cols in lanes): lane L holds col (L&15);
// elements e=0..7 -> K = 8*half + e, e=8..15 -> K = 16 + 8*half + (e-8)
__device__ __forceinline__ v16bf load_frag_b(const bf16* base, int stride, int lane) {
  const int r = lane & 15, hh = lane >> 4;
  const bf16* p = base + (size_t)r * stride + 8 * hh;
  FragU f;
  f.u[0] = *reinterpret_cast<const U4*>(p);
  f.u[1] = *reinterpret_cast<const U4*>(p + 16);
  return f.v;
}

// async 16B global -> LDS copy (ASYNCcnt-tracked, CDNA5)
__device__ __forceinline__ void async_b128(unsigned lds_byte_addr, const void* gsrc) {
  unsigned long long ga = (unsigned long long)gsrc;
  asm volatile("global_load_async_to_lds_b128 %0, %1, off"
               :: "v"(lds_byte_addr), "v"(ga) : "memory");
}
__device__ __forceinline__ void wait_async0() {
  asm volatile("s_wait_asynccnt 0" ::: "memory");
}

// ---------------------------------------------------------------- seg scan
__global__ __launch_bounds__(256) void seg_scan(const int* __restrict__ cid,
                                                int* __restrict__ seg) {
  const int b = blockIdx.x;
  const int tid = threadIdx.x;
  __shared__ int part[256];
  const int PER = S_ / 256;  // 8
  const int base = b * S_ + tid * PER;
  int c[PER];
  int run = 0;
#pragma unroll
  for (int i = 0; i < PER; ++i) {
    int ch = cid[base + i];
    run += ((ch == 32) || (ch == 10)) ? 1 : 0;
    c[i] = run;
  }
  part[tid] = run;
  __syncthreads();
  int acc = run;
  for (int offd = 1; offd < 256; offd <<= 1) {
    int v = (tid >= offd) ? part[tid - offd] : 0;
    __syncthreads();
    acc += v;
    part[tid] = acc;
    __syncthreads();
  }
  const int excl = acc - run;
#pragma unroll
  for (int i = 0; i < PER; ++i) seg[base + i] = excl + c[i];
}

// ---------------------------------------------------------------- x: f32 -> bf16
__global__ __launch_bounds__(256) void cvt_f32_bf16(const float* __restrict__ in,
                                                    bf16* __restrict__ out, int n) {
  const int i = (blockIdx.x * blockDim.x + threadIdx.x) * 4;
  if (i >= n) return;
  F4 f = *reinterpret_cast<const F4*>(in + i);
  B4 o;
  o.h[0] = (bf16)f.x; o.h[1] = (bf16)f.y; o.h[2] = (bf16)f.z; o.h[3] = (bf16)f.w;
  *reinterpret_cast<B4*>(out + i) = o;
}

// ---------------------------------------------------------------- W: f32 [K][N] -> bf16 [N][K]
__global__ __launch_bounds__(256) void cvt_transpose_w(const float* __restrict__ W,
                                                       bf16* __restrict__ WT) {
  __shared__ bf16 tile[32][33];
  const int n0 = blockIdx.x * 32;  // over N3_
  const int k0 = blockIdx.y * 32;  // over E_
  const int tx = threadIdx.x & 31, ty = threadIdx.x >> 5;  // ty 0..7
#pragma unroll
  for (int r = 0; r < 4; ++r) {
    const int k = ty + r * 8;
    tile[k][tx] = (bf16)W[(size_t)(k0 + k) * N3_ + n0 + tx];
  }
  __syncthreads();
#pragma unroll
  for (int r = 0; r < 4; ++r) {
    const int n = ty + r * 8;
    WT[(size_t)(n0 + n) * E_ + k0 + tx] = tile[tx][n];
  }
}

// ---------------------------------------------------------------- QKV GEMM
// C[8192,3072] = A[8192,1024] x BT[3072,1024]^T (+bias; fold 1/sqrt(D) into q,
// 1/D into v; scatter v into vT[h*D+d][B*S+s]). Double-buffered async staging.
__global__ __launch_bounds__(256) void qkv_gemm(const bf16* __restrict__ A,
                                                const bf16* __restrict__ BT,
                                                const float* __restrict__ bias,
                                                bf16* __restrict__ C,
                                                bf16* __restrict__ vT) {
  const int K = E_, N = N3_;
  __shared__ bf16 As[2][128 * 72];  // [m][k], stride 72 (144B, 16B aligned)
  __shared__ bf16 Bs[2][128 * 72];  // [n][k]
  const int tid = threadIdx.x;
  const int lane = tid & 31, wave = tid >> 5;
  const int wr = wave >> 1, wc = wave & 1;  // 4x2 wave grid, 32x64 per wave
  const int r15 = lane & 15, hl = lane >> 4;
  const int m0 = blockIdx.y * 128, n0 = blockIdx.x * 128;

  v8f acc[2][4];
#pragma unroll
  for (int i = 0; i < 2; ++i)
#pragma unroll
    for (int j = 0; j < 4; ++j)
#pragma unroll
      for (int r = 0; r < 8; ++r) acc[i][j][r] = 0.0f;

  const int srow = tid >> 1;        // 0..127
  const int scol = (tid & 1) * 32;  // 0 or 32
  const bf16* arow = A + (size_t)(m0 + srow) * K + scol;
  const bf16* brow = BT + (size_t)(n0 + srow) * K + scol;

  auto stage = [&](int kk, int buf) {
#pragma unroll
    for (int i = 0; i < 4; ++i) {
      async_b128((unsigned)(size_t)&As[buf][srow * 72 + scol + i * 8], arow + kk + i * 8);
      async_b128((unsigned)(size_t)&Bs[buf][srow * 72 + scol + i * 8], brow + kk + i * 8);
    }
  };

  stage(0, 0);
  int cur = 0;
  for (int kk = 0; kk < K; kk += 64) {
    wait_async0();      // current buffer's loads done (next buffer's not yet issued)
    __syncthreads();
    if (kk + 64 < K) stage(kk + 64, cur ^ 1);

#pragma unroll
    for (int ks = 0; ks < 2; ++ks) {
      v16bf af[2], bfv[4];
#pragma unroll
      for (int i = 0; i < 2; ++i)
        af[i] = load_frag_a(&As[cur][(wr * 32 + i * 16) * 72 + ks * 32], 72, lane);
#pragma unroll
      for (int j = 0; j < 4; ++j)
        bfv[j] = load_frag_b(&Bs[cur][(wc * 64 + j * 16) * 72 + ks * 32], 72, lane);
#pragma unroll
      for (int i = 0; i < 2; ++i)
#pragma unroll
        for (int j = 0; j < 4; ++j)
          acc[i][j] = __builtin_amdgcn_wmma_f32_16x16x32_bf16(
              false, af[i], false, bfv[j], (short)0, acc[i][j], false, false);
    }
    cur ^= 1;
  }

#pragma unroll
  for (int i = 0; i < 2; ++i)
#pragma unroll
    for (int j = 0; j < 4; ++j) {
      const int nb = n0 + wc * 64 + j * 16 + r15;
      const float sc = (nb < E_) ? 0.125f : ((nb >= 2 * E_) ? (1.0f / 64.0f) : 1.0f);
      const float bv = bias[nb];
      const bool isv = (nb >= 2 * E_);  // uniform per 16-col tile
#pragma unroll
      for (int r = 0; r < 8; ++r) {
        const int m = m0 + wr * 32 + i * 16 + r + 8 * hl;
        const bf16 val = (bf16)((acc[i][j][r] + bv) * sc);
        C[(size_t)m * N + nb] = val;
        if (isv) vT[(size_t)(nb - 2 * E_) * BS_ + m] = val;
      }
    }
}

// ---------------------------------------------------------------- flash attention
// one block per (128-row q tile, head, batch); 8 waves x 16 q-rows; 64-key chunks;
// double-buffered async K/V staging; visible-chunk walk via monotone segment ids.
__global__ __launch_bounds__(256) void attn_kernel(const bf16* __restrict__ qkv,
                                                   const bf16* __restrict__ vT,
                                                   const int* __restrict__ seg,
                                                   float* __restrict__ out) {
  const int qt = blockIdx.x, head = blockIdx.y, b = blockIdx.z;
  const int tid = threadIdx.x;
  const int lane = tid & 31, wave = tid >> 5;
  const int r15 = lane & 15, hl = lane >> 4;
  const int q0 = qt * QT_;

  __shared__ bf16 Kt[2][64 * 72];  // [kcol][d]
  __shared__ bf16 Vt[2][64 * 72];  // [d][k] (from pre-transposed vT)
  __shared__ bf16 Ps[QT_ * 72];    // [q-row-in-tile][k]

  const bf16* qbase = qkv + (size_t)b * S_ * N3_ + (size_t)head * D_;
  const bf16* kbase = qbase + E_;
  const bf16* vTb = vT + (size_t)head * D_ * BS_ + (size_t)b * S_;  // row d, stride BS_
  const int* segb = seg + b * S_;

  v16bf qf[2];
  {
    const bf16* qrb = qbase + (size_t)(q0 + wave * 16) * N3_;
    qf[0] = load_frag_a(qrb, N3_, lane);
    qf[1] = load_frag_a(qrb + 32, N3_, lane);
  }

  int qrow[8], segq[8];
#pragma unroll
  for (int r = 0; r < 8; ++r) {
    qrow[r] = q0 + wave * 16 + r + 8 * hl;
    segq[r] = segb[qrow[r]];
  }
  const int seg_qmin = segb[q0];
  const int wave_qmax = q0 + wave * 16 + 15;
  const int nchunk = (q0 + QT_) / 64;

  v8f o[4];
  float mrow[8], lrow[8];
#pragma unroll
  for (int t = 0; t < 4; ++t)
#pragma unroll
    for (int r = 0; r < 8; ++r) o[t][r] = 0.0f;
#pragma unroll
  for (int r = 0; r < 8; ++r) { mrow[r] = -1e30f; lrow[r] = 0.0f; }

  auto stage = [&](int j, int buf) {
#pragma unroll
    for (int i = 0; i < 2; ++i) {
      const int c = tid * 2 + i;
      const int row = c >> 3, c8 = (c & 7) * 8;
      async_b128((unsigned)(size_t)&Kt[buf][row * 72 + c8],
                 kbase + (size_t)(j * 64 + row) * N3_ + c8);
      async_b128((unsigned)(size_t)&Vt[buf][row * 72 + c8],
                 vTb + (size_t)row * BS_ + j * 64 + c8);
    }
  };

  // first visible chunk (diagonal chunks are always visible)
  int j = 0;
  while (j < nchunk && segb[j * 64 + 63] < seg_qmin) ++j;
  stage(j, 0);
  int cur = 0;

  while (j < nchunk) {
    int jn = j + 1;
    while (jn < nchunk && segb[jn * 64 + 63] < seg_qmin) ++jn;

    wait_async0();
    __syncthreads();
    if (jn < nchunk) stage(jn, cur ^ 1);  // prefetch next visible chunk

    const bool active = (j * 64) <= wave_qmax;  // wave-uniform causal guard
    if (active) {
      // scores = Qw(16x64) x K^T -> 4 tiles of 16x16
      v8f sacc[4];
#pragma unroll
      for (int t = 0; t < 4; ++t) {
#pragma unroll
        for (int r = 0; r < 8; ++r) sacc[t][r] = 0.0f;
#pragma unroll
        for (int ks = 0; ks < 2; ++ks) {
          v16bf kf = load_frag_b(&Kt[cur][(t * 16) * 72 + ks * 32], 72, lane);
          sacc[t] = __builtin_amdgcn_wmma_f32_16x16x32_bf16(
              false, qf[ks], false, kf, (short)0, sacc[t], false, false);
        }
      }

      // delimiter-causal mask
      int kc[4], segk[4];
#pragma unroll
      for (int t = 0; t < 4; ++t) {
        kc[t] = j * 64 + t * 16 + r15;
        segk[t] = segb[kc[t]];
      }
#pragma unroll
      for (int t = 0; t < 4; ++t)
#pragma unroll
        for (int r = 0; r < 8; ++r) {
          const bool ok = (kc[t] <= qrow[r]) && (segk[t] == segq[r]);
          sacc[t][r] = ok ? sacc[t][r] : -1e9f;
        }

      // online softmax (16-lane row reductions, wave32)
      float mx[8];
#pragma unroll
      for (int r = 0; r < 8; ++r) {
        float m = mrow[r];
#pragma unroll
        for (int t = 0; t < 4; ++t) m = fmaxf(m, sacc[t][r]);
#pragma unroll
        for (int d = 1; d < 16; d <<= 1) m = fmaxf(m, __shfl_xor(m, d, 32));
        mx[r] = m;
      }
      float alpha[8], rsum[8];
#pragma unroll
      for (int r = 0; r < 8; ++r) {
        alpha[r] = __expf(mrow[r] - mx[r]);
        mrow[r] = mx[r];
        rsum[r] = 0.0f;
      }
#pragma unroll
      for (int t = 0; t < 4; ++t)
#pragma unroll
        for (int r = 0; r < 8; ++r) {
          const float p = __expf(sacc[t][r] - mrow[r]);
          rsum[r] += p;
          Ps[(wave * 16 + r + 8 * hl) * 72 + t * 16 + r15] = (bf16)p;  // wave-local
        }
#pragma unroll
      for (int r = 0; r < 8; ++r) {
        float s = rsum[r];
#pragma unroll
        for (int d = 1; d < 16; d <<= 1) s += __shfl_xor(s, d, 32);
        lrow[r] = lrow[r] * alpha[r] + s;
#pragma unroll
        for (int t = 0; t < 4; ++t) o[t][r] *= alpha[r];
      }

      // O += P x V  (V pre-scaled by 1/D)
      v16bf pf[2];
      pf[0] = load_frag_a(&Ps[(wave * 16) * 72], 72, lane);
      pf[1] = load_frag_a(&Ps[(wave * 16) * 72 + 32], 72, lane);
#pragma unroll
      for (int t = 0; t < 4; ++t)
#pragma unroll
        for (int ks = 0; ks < 2; ++ks) {
          v16bf vf = load_frag_b(&Vt[cur][(t * 16) * 72 + ks * 32], 72, lane);
          o[t] = __builtin_amdgcn_wmma_f32_16x16x32_bf16(
              false, pf[ks], false, vf, (short)0, o[t], false, false);
        }
    }
    j = jn;
    cur ^= 1;
  }

#pragma unroll
  for (int t = 0; t < 4; ++t)
#pragma unroll
    for (int r = 0; r < 8; ++r) {
      const int q = qrow[r];
      const int dcol = t * 16 + r15;
      out[((size_t)(b * S_ + q)) * E_ + head * D_ + dcol] = o[t][r] / lrow[r];
    }
}

// ---------------------------------------------------------------- launcher
extern "C" void kernel_launch(void* const* d_in, const int* in_sizes, int n_in,
                              void* d_out, int out_size, void* d_ws, size_t ws_size,
                              hipStream_t stream) {
  (void)in_sizes; (void)n_in; (void)out_size; (void)ws_size;

  const float* x    = (const float*)d_in[0];
  const int*   cid  = (const int*)d_in[1];
  const float* Wqkv = (const float*)d_in[2];
  const float* bqkv = (const float*)d_in[3];
  float* out = (float*)d_out;

  char* ws = (char*)d_ws;
  size_t off = 0;
  bf16* xb   = (bf16*)(ws + off); off += (size_t)BS_ * E_  * sizeof(bf16);  // 16 MB
  bf16* wbT  = (bf16*)(ws + off); off += (size_t)E_ * N3_  * sizeof(bf16);  //  6 MB (as [N][K])
  bf16* qkvb = (bf16*)(ws + off); off += (size_t)BS_ * N3_ * sizeof(bf16);  // 48 MB
  bf16* vTb  = (bf16*)(ws + off); off += (size_t)E_ * BS_  * sizeof(bf16);  // 16 MB
  int*  segp = (int*)(ws + off);  off += (size_t)BS_       * sizeof(int);

  seg_scan<<<B_, 256, 0, stream>>>(cid, segp);

  const int xn = BS_ * E_;
  cvt_f32_bf16<<<xn / 4 / 256, 256, 0, stream>>>(x, xb, xn);
  cvt_transpose_w<<<dim3(N3_ / 32, E_ / 32), 256, 0, stream>>>(Wqkv, wbT);

  qkv_gemm<<<dim3(N3_ / 128, BS_ / 128), 256, 0, stream>>>(xb, wbT, bqkv, qkvb, vTb);

  attn_kernel<<<dim3(S_ / QT_, H_, B_), 256, 0, stream>>>(qkvb, vTb, segp, out);
}
